// BahdanauAttention_48773648614131
// MI455X (gfx1250) — matrix operand order repaired and checked by reading, not compiled, not run on hardware
//
#include <hip/hip_runtime.h>
#include <math.h>

// Problem constants (from reference)
#define B_   128
#define T_   2048
#define D_   512
#define U_   512
#define L_   128
#define HALF 64

// LDS row stride in bf16 elements, padded (+8): 520*2 = 1040 B per row,
// 1040/4 mod 64 = 4 banks of skew per row -> conflict-free ds_load_b128
// across the 16 A-fragment lanes.
#define AS   520

typedef __attribute__((ext_vector_type(16))) __bf16 v16bf;
typedef __attribute__((ext_vector_type(8)))  __bf16 v8bf;
typedef __attribute__((ext_vector_type(8)))  float  v8f;

// ---- dynamic LDS layout for the fused kernel (bytes) ---------------------
#define A_BYTES   (L_ * AS * 2)          // 133120 : bf16 window tile
#define SCB_OFF   (A_BYTES)              // float[512] combined bias row
#define SVW_OFF   (SCB_OFF + 2048)       // float[512] V_w
#define SSC_OFF   (SVW_OFF + 2048)       // float[128] score / weights
#define SRED_OFF  (SSC_OFF + 512)        // float[8]   reductions
#define SMEM_BYTES (SRED_OFF + 32)       // 137760 B total

// ---- workspace layout (bytes) --------------------------------------------
#define W1P_OFF   0                      // bf16[512*512] packed W1
#define W2P_OFF   (512 * 512 * 2)        // bf16[512*512] packed W2
#define QP_OFF    (W2P_OFF + 512 * 512 * 2) // float[128*512] qproj + biases

// ---------------------------------------------------------------------------
// Kernel 0 (x2): pack a [D][U] row-major f32 weight into bf16 laid out as
// WMMA B fragments (16x16x32 bf16): for (nt,k16) tile, lane holds 16
// contiguous bf16 = one 32-byte load.  K order used here (A matches): lanes 0-15 K=0..15,
// lanes 16-31 K=16..31 of the 32-K slab; n = nt*16 + (lane&15).
// ---------------------------------------------------------------------------
__global__ void pack_wmat_kernel(const float* __restrict__ W,
                                 __bf16* __restrict__ Wp)
{
    const int i = blockIdx.x * 256 + threadIdx.x;       // 0 .. 512*512-1
    const int e    =  i        & 15;
    const int lane = (i >> 4)  & 31;
    const int k16  = (i >> 9)  & 15;
    const int nt   =  i >> 13;                          // 0..31
    const int K = k16 * 32 + (lane >> 4) * 16 + e;      // row (D dim)
    const int n = nt * 16 + (lane & 15);                // col (U dim)
    Wp[i] = (__bf16)W[K * U_ + n];
}

// ---------------------------------------------------------------------------
// Kernel 1: batched q_proj GEMM, done ONCE for all batches:
//   QPcb[m][n] = W1_b[n] + W2_b[n] + sum_d query[m][d] * W2_w[d][n]
// 128x512x512 via v_wmma_f32_16x16x32_bf16; one workgroup, 8 waves.
// Replaces a per-block VALU GEMV that would re-read W2_w 128x from L2.
// ---------------------------------------------------------------------------
__global__ __launch_bounds__(256, 1)
void qproj_gemm(const float* __restrict__ query,
                const __bf16* __restrict__ W2p,
                const float* __restrict__ W1_b,
                const float* __restrict__ W2_b,
                float* __restrict__ QPcb)
{
    extern __shared__ char smem[];
    __bf16* sA = (__bf16*)smem;          // 128 x 520 bf16 query tile

    const int tid  = threadIdx.x;
    const int lane = tid & 31;
    const int wave = tid >> 5;

    // stage query (row-major f32) -> LDS bf16
    for (int i = tid; i < B_ * (D_ / 4); i += 256) {
        const int m  = i >> 7;                 // D_/4 == 128
        const int d4 = (i & 127) << 2;
        const float4 v = *(const float4*)(query + (long)m * D_ + d4);
        __bf16* dst = sA + m * AS + d4;
        dst[0] = (__bf16)v.x; dst[1] = (__bf16)v.y;
        dst[2] = (__bf16)v.z; dst[3] = (__bf16)v.w;
    }
    __syncthreads();

    for (int q = 0; q < 4; ++q) {
        const int nt = wave * 4 + q;

        v8f acc[8];
        #pragma unroll
        for (int mt = 0; mt < 8; ++mt)
            #pragma unroll
            for (int e = 0; e < 8; ++e) acc[mt][e] = 0.f;

        const __bf16* arow_base = sA + (lane & 15) * AS + ((lane >> 4) * 16);

        for (int k16 = 0; k16 < 16; ++k16) {
            const v16bf bfrag =
                *(const v16bf*)(W2p + ((size_t)(nt * 16 + k16) * 32 + lane) * 16);
            #pragma unroll
            for (int mt = 0; mt < 8; ++mt) {
                const __bf16* ap = arow_base + mt * 16 * AS + k16 * 32;
                const v8bf lo = *(const v8bf*)(ap);
                const v8bf hi = *(const v8bf*)(ap + 8);
                const v16bf afrag = __builtin_shufflevector(
                    lo, hi, 0,1,2,3,4,5,6,7,8,9,10,11,12,13,14,15);
                acc[mt] = __builtin_amdgcn_wmma_f32_16x16x32_bf16(
                    false, afrag, false, bfrag, (short)0, acc[mt], false, false);
            }
        }

        // epilogue: add biases, scatter to QPcb (C layout: VGPR r -> M=r(+8))
        const int   n    = nt * 16 + (lane & 15);
        const float bias = W1_b[n] + W2_b[n];
        #pragma unroll
        for (int mt = 0; mt < 8; ++mt) {
            #pragma unroll
            for (int r = 0; r < 8; ++r) {
                const int m = mt * 16 + (lane >> 4) * 8 + r;
                QPcb[(long)m * U_ + n] = acc[mt][r] + bias;
            }
        }
    }
}

// ---------------------------------------------------------------------------
// Kernel 2: fully fused Bahdanau attention, one workgroup per batch element.
// ---------------------------------------------------------------------------
__global__ __launch_bounds__(256, 1)
void bahdanau_fused(const float* __restrict__ values,
                    const int*   __restrict__ pos,
                    const float* __restrict__ Vw,
                    const __bf16* __restrict__ W1p,
                    const float* __restrict__ QPcb,
                    float* __restrict__ ctx_out,
                    float* __restrict__ attn_out)
{
    extern __shared__ char smem[];
    __bf16* sA     = (__bf16*)smem;
    float*  sCB    = (float*)(smem + SCB_OFF);
    float*  sVw    = (float*)(smem + SVW_OFF);
    float*  sScore = (float*)(smem + SSC_OFF);
    float*  sRed   = (float*)(smem + SRED_OFF);

    const int b    = blockIdx.x;
    const int tid  = threadIdx.x;
    const int lane = tid & 31;
    const int wave = tid >> 5;

    // ---- stage combined bias row / V_w / score init ----------------------
    sCB[tid]       = QPcb[(long)b * U_ + tid];
    sCB[tid + 256] = QPcb[(long)b * U_ + tid + 256];
    sVw[tid]       = Vw[tid];
    sVw[tid + 256] = Vw[tid + 256];
    if (tid < L_) sScore[tid] = 0.0f;

    // ---- gather window rows into LDS as bf16 (zero outside [start,end)) --
    const int p     = pos[b];
    const int start = max(p - HALF, 0);
    const int end   = min(p + HALF, T_);
    for (int i = tid; i < L_ * (D_ / 4); i += 256) {
        const int l  = i >> 7;                // D_/4 == 128
        const int d4 = (i & 127) << 2;
        const int t  = start + l;
        float4 v = make_float4(0.f, 0.f, 0.f, 0.f);
        if (t < end)
            v = *(const float4*)(values + ((long)t * B_ + b) * D_ + d4);
        __bf16* dst = sA + l * AS + d4;
        dst[0] = (__bf16)v.x; dst[1] = (__bf16)v.y;
        dst[2] = (__bf16)v.z; dst[3] = (__bf16)v.w;
    }
    __syncthreads();

    // ---- main GEMM: hidden = windows @ W1 via v_wmma_f32_16x16x32_bf16 ---
    // 8 waves x 4 N-tiles cover U=512; 8 M-tile accumulators per wave so
    // every B fragment (L2-resident packed W1) is reused 8x.
    for (int q = 0; q < 4; ++q) {
        const int nt = wave * 4 + q;

        v8f acc[8];
        #pragma unroll
        for (int mt = 0; mt < 8; ++mt)
            #pragma unroll
            for (int e = 0; e < 8; ++e) acc[mt][e] = 0.f;

        const __bf16* arow_base = sA + (lane & 15) * AS + ((lane >> 4) * 16);

        for (int k16 = 0; k16 < 16; ++k16) {
            const v16bf bfrag =
                *(const v16bf*)(W1p + ((size_t)(nt * 16 + k16) * 32 + lane) * 16);
            #pragma unroll
            for (int mt = 0; mt < 8; ++mt) {
                const __bf16* ap = arow_base + mt * 16 * AS + k16 * 32;
                const v8bf lo = *(const v8bf*)(ap);        // K = khalf..+7
                const v8bf hi = *(const v8bf*)(ap + 8);    // K = khalf+8..+15
                const v16bf afrag = __builtin_shufflevector(
                    lo, hi, 0,1,2,3,4,5,6,7,8,9,10,11,12,13,14,15);
                acc[mt] = __builtin_amdgcn_wmma_f32_16x16x32_bf16(
                    false, afrag, false, bfrag, (short)0, acc[mt], false, false);
            }
        }

        // ---- fused epilogue: score[m] += sum_n tanh(h + cb[n]) * Vw[n] ---
        const int   n  = nt * 16 + (lane & 15);
        const float cb = sCB[n];
        const float vw = sVw[n];
        #pragma unroll
        for (int mt = 0; mt < 8; ++mt) {
            #pragma unroll
            for (int r = 0; r < 8; ++r) {
                float h = tanhf(acc[mt][r] + cb) * vw;
                h += __shfl_xor(h, 8, 16);
                h += __shfl_xor(h, 4, 16);
                h += __shfl_xor(h, 2, 16);
                h += __shfl_xor(h, 1, 16);
                if ((lane & 15) == 0)
                    atomicAdd(&sScore[mt * 16 + (lane >> 4) * 8 + r], h);
            }
        }
    }
    __syncthreads();

    // ---- softmax over L=128 (V_b is softmax-invariant: dropped) ----------
    if (tid < L_) {
        float m = sScore[tid];
        m = fmaxf(m, __shfl_xor(m, 16, 32));
        m = fmaxf(m, __shfl_xor(m,  8, 32));
        m = fmaxf(m, __shfl_xor(m,  4, 32));
        m = fmaxf(m, __shfl_xor(m,  2, 32));
        m = fmaxf(m, __shfl_xor(m,  1, 32));
        if (lane == 0) sRed[wave] = m;
    }
    __syncthreads();
    if (tid < L_) {
        const float gmax = fmaxf(fmaxf(sRed[0], sRed[1]),
                                 fmaxf(sRed[2], sRed[3]));
        const float e = expf(sScore[tid] - gmax);
        float s = e;
        s += __shfl_xor(s, 16, 32);
        s += __shfl_xor(s,  8, 32);
        s += __shfl_xor(s,  4, 32);
        s += __shfl_xor(s,  2, 32);
        s += __shfl_xor(s,  1, 32);
        if (lane == 0) sRed[4 + wave] = s;
        sScore[tid] = e;
    }
    __syncthreads();
    if (tid < L_) {
        const float tot = sRed[4] + sRed[5] + sRed[6] + sRed[7];
        const float w = sScore[tid] / tot;
        sScore[tid] = w;
        attn_out[b * L_ + tid] = w;
    }
    __syncthreads();

    // ---- context[d] = sum_l w[l] * window[l][d] --------------------------
    for (int d = tid; d < D_; d += 256) {
        float acc = 0.f;
        for (int l = 0; l < L_; ++l)
            acc = fmaf(sScore[l], (float)sA[l * AS + d], acc);
        ctx_out[b * D_ + d] = acc;
    }
}

// ---------------------------------------------------------------------------
extern "C" void kernel_launch(void* const* d_in, const int* in_sizes, int n_in,
                              void* d_out, int out_size, void* d_ws, size_t ws_size,
                              hipStream_t stream) {
    (void)in_sizes; (void)n_in; (void)out_size; (void)ws_size;
    const float* query  = (const float*)d_in[0];
    const float* values = (const float*)d_in[1];
    const int*   pos    = (const int*)  d_in[2];
    const float* W1_w   = (const float*)d_in[3];
    const float* W1_b   = (const float*)d_in[4];
    const float* W2_w   = (const float*)d_in[5];
    const float* W2_b   = (const float*)d_in[6];
    const float* V_w    = (const float*)d_in[7];
    // d_in[8] = V_b : softmax-invariant, intentionally unused.

    float* ctx  = (float*)d_out;            // [B, D]
    float* attn = ctx + (size_t)B_ * D_;    // [B, L, 1]

    char* ws = (char*)d_ws;
    __bf16* W1p  = (__bf16*)(ws + W1P_OFF); // 512 KB packed W1 (bf16)
    __bf16* W2p  = (__bf16*)(ws + W2P_OFF); // 512 KB packed W2 (bf16)
    float*  QPcb = (float*)(ws + QP_OFF);   // 256 KB qproj + biases (f32)

    // Allow >64KB dynamic LDS (host-side attribute set; capture-safe).
    (void)hipFuncSetAttribute((const void*)bahdanau_fused,
                              hipFuncAttributeMaxDynamicSharedMemorySize,
                              SMEM_BYTES);
    (void)hipFuncSetAttribute((const void*)qproj_gemm,
                              hipFuncAttributeMaxDynamicSharedMemorySize,
                              A_BYTES);

    pack_wmat_kernel<<<(D_ * U_) / 256, 256, 0, stream>>>(W1_w, W1p);
    pack_wmat_kernel<<<(D_ * U_) / 256, 256, 0, stream>>>(W2_w, W2p);
    qproj_gemm<<<1, 256, A_BYTES, stream>>>(query, W2p, W1_b, W2_b, QPcb);
    bahdanau_fused<<<B_, 256, SMEM_BYTES, stream>>>(
        values, pos, V_w, W1p, QPcb, ctx, attn);
}
